// LinearAttention_33526514713266
// MI455X (gfx1250) — matrix operand order repaired and checked by reading, not compiled, not run on hardware
//
#include <hip/hip_runtime.h>
#include <stdint.h>

// ---------------------------------------------------------------------------
// CDNA5 (gfx1250) linear-attention pipeline, bf16 WMMA w/ f32 accumulation.
// Async global->LDS (ASYNCcnt) double-buffered staging; zero-VGPR copies.
// ---------------------------------------------------------------------------

typedef __attribute__((ext_vector_type(16))) __bf16 bf16x16;
typedef __attribute__((ext_vector_type(8)))  float  f32x8;
typedef __attribute__((ext_vector_type(4)))  int    i32x4;

#if __has_builtin(__builtin_amdgcn_global_load_async_to_lds_b128)
#define HAVE_ASYNC_LDS 1
#else
#define HAVE_ASYNC_LDS 0
#endif

__device__ __forceinline__ unsigned short f2bf(float f) {
  unsigned int u = __float_as_uint(f);
  u += 0x7fffu + ((u >> 16) & 1u);   // round-to-nearest-even
  return (unsigned short)(u >> 16);
}

// Pack two f32 -> two bf16 (one dword). Prefer hardware v_cvt_pk_bf16_f32.
__device__ __forceinline__ unsigned int pack2_bf16(float a, float b) {
#if __has_builtin(__builtin_amdgcn_cvt_pk_bf16_f32)
  typedef __attribute__((ext_vector_type(2))) __bf16 bf16x2;
  bf16x2 r = __builtin_amdgcn_cvt_pk_bf16_f32(a, b);
  return __builtin_bit_cast(unsigned int, r);
#else
  return (unsigned int)f2bf(a) | ((unsigned int)f2bf(b) << 16);
#endif
}

// 16-byte global -> LDS copy. Async DMA (no VGPR data) when available.
__device__ __forceinline__ void async_cp16(unsigned short* lds_dst,
                                           const unsigned short* gsrc) {
#if HAVE_ASYNC_LDS
  __builtin_amdgcn_global_load_async_to_lds_b128(
      (__attribute__((address_space(1))) i32x4*)(uintptr_t)gsrc,
      (__attribute__((address_space(3))) i32x4*)(unsigned int)(uintptr_t)lds_dst,
      0, 0);
#else
  *reinterpret_cast<uint4*>(lds_dst) = *reinterpret_cast<const uint4*>(gsrc);
#endif
}

__device__ __forceinline__ void wait_async() {
#if HAVE_ASYNC_LDS
#if __has_builtin(__builtin_amdgcn_s_wait_asynccnt)
  __builtin_amdgcn_s_wait_asynccnt(0);
#else
  asm volatile("s_wait_asynccnt 0" ::: "memory");
#endif
#endif
}

union FragAB {
  bf16x16 v;
  uint4   q[2];
};

// LDS row stride (bf16 elems): 48 -> 96 B/row, 16-byte aligned rows.
#define LDSS 48

// ---------------------------------------------------------------------------
// Tiled GEMM:  O[M x N] = A[M x K] * B   (all bf16 in, f32 accumulate)
//   A: row-major M x K (lda = K), staged via async global->LDS.
//   B_NK = true : B supplied N x K row-major (ldb = K) -> async direct stage.
//   B_NK = false: B supplied K x N row-major (ldb = N) -> VGPR transpose stage.
// 256 threads (8 wave32), 128x128 block tile, K-chunk 32, double-buffered LDS.
// Wave grid 2x4: each wave = 64x32 tile = 4x2 v_wmma_f32_16x16x32_bf16.
// ---------------------------------------------------------------------------
template <bool B_NK, bool OUT_BF16>
__global__ __launch_bounds__(256) void gemm_wmma(
    const unsigned short* __restrict__ Ap, const unsigned short* __restrict__ Bp,
    void* __restrict__ Op, int K, int ldb, int ldo,
    long strideA, long strideB, long strideO) {
  __shared__ unsigned short As[2][128 * LDSS];
  __shared__ unsigned short Bs[2][128 * LDSS];

  const int tid  = threadIdx.x;
  const int lane = tid & 31;
  const int wave = tid >> 5;
  const int wm   = wave >> 2;   // 0..1  -> 64-row wave tile
  const int wn   = wave & 3;    // 0..3  -> 32-col wave tile
  const int lh   = lane >> 4;   // half-wave select
  const int l16  = lane & 15;

  const int m0  = blockIdx.y * 128;
  const int n0  = blockIdx.x * 128;
  const int lda = K;

  const unsigned short* A = Ap + (long)blockIdx.z * strideA;
  const unsigned short* B = Bp + (long)blockIdx.z * strideB;
  const long zO = (long)blockIdx.z * strideO;

  // ---- A tile: 128x32, two 16B chunks per thread, direct copy -------------
  auto stage_A = [&](int buf, int kc) {
#pragma unroll
    for (int i = 0; i < 2; ++i) {
      const int lin = tid + (i << 8);              // 512 uint4s (8 bf16 each)
      const int row = lin >> 2, q = lin & 3;       // 4 chunks per 32-elem row
      async_cp16(&As[buf][row * LDSS + q * 8],
                 A + (long)(m0 + row) * lda + kc + q * 8);
    }
  };

  // ---- B tile, N x K source: direct copy (async) --------------------------
  auto stage_B_nk = [&](int buf, int kc) {
#pragma unroll
    for (int i = 0; i < 2; ++i) {
      const int lin = tid + (i << 8);
      const int row = lin >> 2, q = lin & 3;       // row = n index
      async_cp16(&Bs[buf][row * LDSS + q * 8],
                 B + (long)(n0 + row) * ldb + kc + q * 8);
    }
  };

  // ---- B tile, K x N source: VGPR transpose stage (8 VGPRs in flight) -----
  uint4 br[2];
  auto load_B_regs = [&](int kc) {
#pragma unroll
    for (int i = 0; i < 2; ++i) {
      const int lin = tid + (i << 8);
      const int kr = lin >> 4, q = lin & 15;       // 16 chunks per 128-col row
      br[i] = *reinterpret_cast<const uint4*>(B + (long)(kc + kr) * ldb + n0 + q * 8);
    }
  };
  auto store_B_lds = [&](int buf) {
#pragma unroll
    for (int i = 0; i < 2; ++i) {
      const int lin = tid + (i << 8);
      const int kr = lin >> 4, c0 = (lin & 15) * 8;
      unsigned short* p = &Bs[buf][c0 * LDSS + kr];
      p[0 * LDSS] = (unsigned short)(br[i].x);
      p[1 * LDSS] = (unsigned short)(br[i].x >> 16);
      p[2 * LDSS] = (unsigned short)(br[i].y);
      p[3 * LDSS] = (unsigned short)(br[i].y >> 16);
      p[4 * LDSS] = (unsigned short)(br[i].z);
      p[5 * LDSS] = (unsigned short)(br[i].z >> 16);
      p[6 * LDSS] = (unsigned short)(br[i].w);
      p[7 * LDSS] = (unsigned short)(br[i].w >> 16);
    }
  };

  f32x8 acc[4][2] = {};
  const int nk = K >> 5;

  // prologue: stage chunk 0 into buffer 0
  stage_A(0, 0);
  if constexpr (B_NK) {
    stage_B_nk(0, 0);
  } else {
    load_B_regs(0);
    store_B_lds(0);
  }
  wait_async();
  __syncthreads();

  for (int kt = 0; kt < nk; ++kt) {
    const int  cur  = kt & 1, nxt = cur ^ 1;
    const bool more = (kt + 1 < nk);

    // kick off chunk k+1 while chunk k computes (safe: LDS[nxt] was last read
    // in iteration k-1, before the previous barrier)
    if (more) {
      const int kc = (kt + 1) << 5;
      stage_A(nxt, kc);
      if constexpr (B_NK) stage_B_nk(nxt, kc);
      else                load_B_regs(kc);
    }

    // ---- fragments + 8 WMMAs off LDS[cur] --------------------------------
    FragAB bfr[2];
#pragma unroll
    for (int tn = 0; tn < 2; ++tn) {
      const int nr = wn * 32 + tn * 16 + l16;
      bfr[tn].q[0] = *reinterpret_cast<const uint4*>(&Bs[cur][nr * LDSS + lh * 8]);
      bfr[tn].q[1] = *reinterpret_cast<const uint4*>(&Bs[cur][nr * LDSS + 16 + lh * 8]);
    }
#pragma unroll
    for (int tm = 0; tm < 4; ++tm) {
      const int mr = wm * 64 + tm * 16 + l16;
      FragAB afr;
      afr.q[0] = *reinterpret_cast<const uint4*>(&As[cur][mr * LDSS + lh * 8]);
      afr.q[1] = *reinterpret_cast<const uint4*>(&As[cur][mr * LDSS + 16 + lh * 8]);
#pragma unroll
      for (int tn = 0; tn < 2; ++tn) {
        acc[tm][tn] = __builtin_amdgcn_wmma_f32_16x16x32_bf16(
            false, afr.v, false, bfr[tn].v, (short)0, acc[tm][tn], false, false);
      }
    }

    if (more) {
      if constexpr (!B_NK) store_B_lds(nxt);
    }
    wait_async();        // next chunk's async copies done before the barrier
    __syncthreads();
  }

  // ---- epilogue: C/D layout VGPR r -> row r+8*lh, col l16 -----------------
#pragma unroll
  for (int tm = 0; tm < 4; ++tm) {
#pragma unroll
    for (int tn = 0; tn < 2; ++tn) {
      const int col = n0 + wn * 32 + tn * 16 + l16;
#pragma unroll
      for (int r = 0; r < 8; ++r) {
        const int  row = m0 + wm * 64 + tm * 16 + lh * 8 + r;
        const long o   = zO + (long)row * ldo + col;
        if constexpr (OUT_BF16) {
          reinterpret_cast<unsigned short*>(Op)[o] = f2bf(acc[tm][tn][r]);
        } else {
          reinterpret_cast<float*>(Op)[o] = acc[tm][tn][r];
        }
      }
    }
  }
}

// ---------------------------------------------------------------------------
// f32 -> bf16 bulk conversion (bandwidth bound; v_cvt_pk_bf16_f32).
// ---------------------------------------------------------------------------
__global__ __launch_bounds__(256) void f32_to_bf16(const float* __restrict__ src,
                                                   unsigned short* __restrict__ dst,
                                                   int n4) {
  const int stride = gridDim.x * blockDim.x;
  for (int i = blockIdx.x * blockDim.x + threadIdx.x; i < n4; i += stride) {
    const float4 v = reinterpret_cast<const float4*>(src)[i];
    uint2 o;
    o.x = pack2_bf16(v.x, v.y);
    o.y = pack2_bf16(v.z, v.w);
    reinterpret_cast<uint2*>(dst)[i] = o;
  }
}

// ---------------------------------------------------------------------------
// Softmax over N (rows of q). One block per (b, c) row; row cached in LDS.
// ---------------------------------------------------------------------------
__global__ __launch_bounds__(256) void softmax_rows(const float* __restrict__ qk,
                                                    unsigned short* __restrict__ qs) {
  __shared__ float row[4096];
  __shared__ float red[256];
  const int tid = threadIdx.x;
  const int c = blockIdx.x, b = blockIdx.y;
  const float* src = qk + ((long)b * 1024 + c) * 4096;
  unsigned short* dst = qs + ((long)b * 512 + c) * 4096;

  float lmax = -3.4e38f;
#pragma unroll
  for (int i = 0; i < 16; ++i) {
    const float v = src[tid + (i << 8)];
    row[tid + (i << 8)] = v;
    lmax = fmaxf(lmax, v);
  }
  red[tid] = lmax;
  __syncthreads();
  for (int s = 128; s > 0; s >>= 1) {
    if (tid < s) red[tid] = fmaxf(red[tid], red[tid + s]);
    __syncthreads();
  }
  const float m = red[0];
  __syncthreads();

  float ls = 0.f;
#pragma unroll
  for (int i = 0; i < 16; ++i) {
    const float e = __expf(row[tid + (i << 8)] - m);
    row[tid + (i << 8)] = e;
    ls += e;
  }
  red[tid] = ls;
  __syncthreads();
  for (int s = 128; s > 0; s >>= 1) {
    if (tid < s) red[tid] += red[tid + s];
    __syncthreads();
  }
  const float inv = 1.0f / red[0];
#pragma unroll
  for (int i = 0; i < 16; ++i)
    dst[tid + (i << 8)] = f2bf(row[tid + (i << 8)] * inv);
}

// ---------------------------------------------------------------------------
// Softmax over C (columns of k). Thread = one column n; coalesced across n.
// ---------------------------------------------------------------------------
__global__ __launch_bounds__(256) void softmax_cols(const float* __restrict__ qk,
                                                    unsigned short* __restrict__ ks) {
  const int n = blockIdx.x * 256 + threadIdx.x;
  const int b = blockIdx.y;
  const float* src = qk + ((long)b * 1024 + 512) * 4096 + n;  // k block rows
  unsigned short* dst = ks + ((long)b * 512) * 4096 + n;

  float m = -3.4e38f, s = 0.f;
  for (int c = 0; c < 512; ++c) {
    const float v  = src[(long)c * 4096];
    const float nm = fmaxf(m, v);
    s = s * __expf(m - nm) + __expf(v - nm);
    m = nm;
  }
  const float inv = 1.0f / s;
  for (int c = 0; c < 512; ++c) {
    const float v = src[(long)c * 4096];
    dst[(long)c * 4096] = f2bf(__expf(v - m) * inv);
  }
}

// ---------------------------------------------------------------------------
// Orchestration:
//   xb = bf16(x); wb = bf16(w_qkv); pb = bf16(w_proj)
//   qk = wb[0:1024] @ xb (f32, for softmax)     v = bf16(wb[1024:] @ xb)
//   q = softmax_n(q) -> bf16 ;  k = softmax_c(k) -> bf16
//   ctx = bf16(k @ v^T) ;  M = bf16(pb @ ctx^T) ;  out = M @ q  (f32)
// ---------------------------------------------------------------------------
extern "C" void kernel_launch(void* const* d_in, const int* in_sizes, int n_in,
                              void* d_out, int out_size, void* d_ws, size_t ws_size,
                              hipStream_t stream) {
  (void)in_sizes; (void)n_in; (void)out_size; (void)ws_size;
  const float* x      = (const float*)d_in[0];   // (8, 512, 64, 64)
  const float* w_qkv  = (const float*)d_in[1];   // (1536, 512)
  const float* w_proj = (const float*)d_in[2];   // (512, 512)
  float* out = (float*)d_out;                    // (8, 512, 64, 64) f32

  constexpr int Bn = 8, C = 512, N = 4096;

  char* ws = (char*)d_ws;
  size_t off = 0;
  float* qk = (float*)(ws + off);              off += (size_t)Bn * 1024 * N * 4;
  unsigned short* xb     = (unsigned short*)(ws + off); off += (size_t)Bn * C * N * 2;
  unsigned short* wqkvb  = (unsigned short*)(ws + off); off += (size_t)3 * C * C * 2;
  unsigned short* wprojb = (unsigned short*)(ws + off); off += (size_t)C * C * 2;
  unsigned short* vb     = (unsigned short*)(ws + off); off += (size_t)Bn * C * N * 2;
  unsigned short* qs     = (unsigned short*)(ws + off); off += (size_t)Bn * C * N * 2;
  unsigned short* ks     = (unsigned short*)(ws + off); off += (size_t)Bn * C * N * 2;
  unsigned short* ctxb   = (unsigned short*)(ws + off); off += (size_t)Bn * C * C * 2;
  unsigned short* Mb     = (unsigned short*)(ws + off);

  dim3 blk(256);

  // bulk bf16 conversions (one pass each)
  f32_to_bf16<<<dim3(2048), blk, 0, stream>>>(x, xb, Bn * C * N / 4);
  f32_to_bf16<<<dim3(96),   blk, 0, stream>>>(w_qkv, wqkvb, 3 * C * C / 4);
  f32_to_bf16<<<dim3(32),   blk, 0, stream>>>(w_proj, wprojb, C * C / 4);

  // qk = w_qkv[0:1024] @ x   (f32 out for softmax precision)
  gemm_wmma<false, false><<<dim3(N / 128, 1024 / 128, Bn), blk, 0, stream>>>(
      wqkvb, xb, qk, /*K=*/C, /*ldb=*/N, /*ldo=*/N, 0L, (long)C * N, (long)1024 * N);

  // v = bf16(w_qkv[1024:1536] @ x)
  gemm_wmma<false, true><<<dim3(N / 128, C / 128, Bn), blk, 0, stream>>>(
      wqkvb + (size_t)1024 * C, xb, vb, C, N, N, 0L, (long)C * N, (long)C * N);

  softmax_rows<<<dim3(C, Bn), blk, 0, stream>>>(qk, qs);
  softmax_cols<<<dim3(N / 256, Bn), blk, 0, stream>>>(qk, ks);

  // ctx = bf16(ks @ vb^T)   (B supplied N x K -> async direct staging)
  gemm_wmma<true, true><<<dim3(C / 128, C / 128, Bn), blk, 0, stream>>>(
      ks, vb, ctxb, /*K=*/N, /*ldb=*/N, /*ldo=*/C, (long)C * N, (long)C * N, (long)C * C);

  // Mb = bf16(w_proj @ ctx^T)
  gemm_wmma<true, true><<<dim3(C / 128, C / 128, Bn), blk, 0, stream>>>(
      wprojb, ctxb, Mb, C, C, C, 0L, (long)C * C, (long)C * C);

  // out = Mb @ qs  (f32 straight to d_out)
  gemm_wmma<false, false><<<dim3(N / 128, C / 128, Bn), blk, 0, stream>>>(
      Mb, qs, out, C, N, N, (long)C * C, (long)C * N, (long)C * N);
}
